// Reasoner_50525995270777
// MI455X (gfx1250) — matrix operand order repaired and checked by reading, not compiled, not run on hardware
//
#include <hip/hip_runtime.h>

typedef __bf16 bf16_t;
typedef bf16_t v16bf __attribute__((ext_vector_type(16)));
typedef float  v8f   __attribute__((ext_vector_type(8)));
typedef unsigned int v8u __attribute__((ext_vector_type(8)));
typedef unsigned short u16;
typedef int gv4i __attribute__((vector_size(16)));   // matches builtin param type

#define FEAT 256
#define LSEQ 8
#define NGRP 4096
#define BCH  32768            // total chains
#define RT_COUNT 2048         // BCH/16 row tiles
#define SSTR 260              // padded f32 LDS row stride (floats)
#define STRB 264              // padded bf16 LDS row stride (ushorts), 528 B rows (16B aligned)
#define WMAT_ELEMS 65536      // 256*256

// workspace layout (bytes)
#define WB_OFF   ((size_t)0)                                  // 5 bf16 weight mats = 655360 B
#define A_OFF    ((size_t)655360)
#define A_BYTES  ((size_t)7 * RT_COUNT * 16 * 32 * 16)        // 117440512 B (bf16 frag layout)
#define Z_OFF    (A_OFF + A_BYTES)
#define Z_BYTES  ((size_t)8 * RT_COUNT * 16 * 32 * 16)        // 134217728 B
#define CUM_OFF  (Z_OFF + Z_BYTES)
#define SSEL_OFF (CUM_OFF + (size_t)BCH * 4)

#if __has_builtin(__builtin_amdgcn_global_load_async_to_lds_b128)
#define HAVE_ASYNC 1
#else
#define HAVE_ASYNC 0
#endif

// ---------- helpers ----------
__device__ __forceinline__ unsigned pack_bf16(float lo, float hi) {
  unsigned a = __float_as_uint(lo);
  unsigned b = __float_as_uint(hi);
  a += 0x7FFFu + ((a >> 16) & 1u);       // round-to-nearest-even
  b += 0x7FFFu + ((b >> 16) & 1u);
  return (a >> 16) | (b & 0xFFFF0000u);
}
__device__ __forceinline__ u16 bf16_of(float f) {
  unsigned a = __float_as_uint(f);
  a += 0x7FFFu + ((a >> 16) & 1u);
  return (u16)(a >> 16);
}
__device__ __forceinline__ float bf_lo(unsigned u) { return __uint_as_float(u << 16); }
__device__ __forceinline__ float bf_hi(unsigned u) { return __uint_as_float(u & 0xFFFF0000u); }

#if HAVE_ASYNC
// one 16-B async global->LDS transfer per lane (ASYNCcnt-tracked DMA)
__device__ __forceinline__ void async_copy16(const void* g, void* l) {
  __builtin_amdgcn_global_load_async_to_lds_b128(
      (__attribute__((address_space(1))) gv4i*)(uintptr_t)g,
      (__attribute__((address_space(3))) gv4i*)l,
      0, 0);
}
__device__ __forceinline__ void async_wait_all() {
#if __has_builtin(__builtin_amdgcn_s_wait_asynccnt)
  __builtin_amdgcn_s_wait_asynccnt(0);
#else
  asm volatile("s_wait_asynccnt 0x0" ::: "memory");
#endif
}
#endif

// A-fragment straight from a bf16 LDS row: two ds_load_b128, no VALU.
// lane<16 -> M=lane, K = k0..k0+7 and k0+16..k0+23 (k0 = kc*32); lane>=16 -> k0 += 8.
__device__ __forceinline__ v16bf frag_from_lds_bf(const u16* row, int k0) {
  uint4 q0 = *(const uint4*)(row + k0);
  uint4 q1 = *(const uint4*)(row + k0 + 16);
  v8u u = {q0.x, q0.y, q0.z, q0.w, q1.x, q1.y, q1.z, q1.w};
  return __builtin_bit_cast(v16bf, u);
}

// B-fragment (32x16 bf16): lane -> column n, lanes<16 hold K=k0..k0+7,k0+16..k0+23; lanes>=16 K=k0+8..k0+15,k0+24..k0+31.
// B[k][n] = W[n][k] (row-major bf16 weight, two 8-bf16 runs 16 apart = two 16-B loads)
__device__ __forceinline__ v16bf load_b_frag(const u16* __restrict__ W, int n, int k0) {
  const uint4* p = (const uint4*)(W + n * FEAT + k0 - ((k0 & 31) >> 1));
  uint4 q0 = p[0], q1 = p[2];
  v8u u = {q0.x, q0.y, q0.z, q0.w, q1.x, q1.y, q1.z, q1.w};
  return __builtin_bit_cast(v16bf, u);
}

__device__ __forceinline__ v8f wmma_bf(v16bf a, v16bf b, v8f c) {
  return __builtin_amdgcn_wmma_f32_16x16x32_bf16(false, a, false, b, (short)0, c, false, false);
}

// ---------- kernel 0: weights f32 -> bf16 ----------
__global__ void __launch_bounds__(256) prep_weights(
    const float* __restrict__ Wc, const float* __restrict__ Wcs,
    const float* __restrict__ Wanti, u16* __restrict__ WB) {
  int idx = blockIdx.x * 256 + threadIdx.x;
  if (idx >= 5 * WMAT_ELEMS) return;
  int mat = idx >> 16;
  int r   = (idx >> 8) & 255;
  int k   = idx & 255;
  float v;
  if      (mat == 0) v = Wc[r * 512 + k];            // Wc1
  else if (mat == 1) v = Wc[r * 512 + 256 + k];      // Wc2
  else if (mat == 2) v = Wcs[r * 256 + k];           // Wcs
  else if (mat == 3) v = Wanti[r * 512 + k];         // Wa1
  else               v = Wanti[r * 512 + 256 + k];   // Wa2
  WB[idx] = bf16_of(v);
}

// ---------- kernel 1: A[t] and Z[l] GEMMs, output in C-fragment bf16 layout ----------
__global__ void __launch_bounds__(128) precompute_kernel(
    const float* __restrict__ x, const u16* __restrict__ WB,
    uint4* __restrict__ Afrag, uint4* __restrict__ Zfrag) {
#if HAVE_ASYNC
  __shared__ float xs[2][16 * SSTR];   // f32 async staging
#endif
  __shared__ u16 xbf[2][16 * STRB];    // bf16 operand image
  const int rt  = blockIdx.x;
  const int job = blockIdx.y;          // 0..7 -> Z[l]; 8..14 -> A[t]
  const bool isA = job >= LSEQ;
  const int l0 = isA ? (job - LSEQ) : job;
  const int tid = threadIdx.x;
  const int rb = rt * 16;
  const int npass = isA ? 2 : 1;

#if HAVE_ASYNC
  for (int p = 0; p < npass; ++p)
    for (int idx = tid; idx < 16 * 64; idx += 128) {
      int m = idx >> 6, c = (idx & 63) * 4;
      async_copy16(x + ((size_t)(rb + m) * LSEQ + l0 + p) * FEAT + c, &xs[p][m * SSTR + c]);
    }
  async_wait_all();
  __syncthreads();
  for (int p = 0; p < npass; ++p)
    for (int idx = tid; idx < 16 * 64; idx += 128) {
      int m = idx >> 6, c = (idx & 63) * 4;
      const float* src = &xs[p][m * SSTR + c];
      uint2 o; o.x = pack_bf16(src[0], src[1]); o.y = pack_bf16(src[2], src[3]);
      *(uint2*)(&xbf[p][m * STRB + c]) = o;
    }
#else
  for (int p = 0; p < npass; ++p)
    for (int idx = tid; idx < 16 * 64; idx += 128) {
      int m = idx >> 6, c = (idx & 63) * 4;
      float4 v = *(const float4*)(x + ((size_t)(rb + m) * LSEQ + l0 + p) * FEAT + c);
      uint2 o; o.x = pack_bf16(v.x, v.y); o.y = pack_bf16(v.z, v.w);
      *(uint2*)(&xbf[p][m * STRB + c]) = o;
    }
#endif
  __syncthreads();

  const int w = tid >> 5, lane = tid & 31, ln = lane & 15, hi = lane >> 4;
  const u16* Wp0 = WB + (isA ? 3 : 1) * WMAT_ELEMS;  // Wa1 or Wc2
  const u16* Wp1 = WB + 4 * WMAT_ELEMS;              // Wa2

  v8f zero = {0.f, 0.f, 0.f, 0.f, 0.f, 0.f, 0.f, 0.f};
  v8f acc[4];
#pragma unroll
  for (int i = 0; i < 4; ++i) acc[i] = zero;

  for (int p = 0; p < npass; ++p) {
    const u16* xrow = &xbf[p][ln * STRB];
    const u16* W = p ? Wp1 : Wp0;
#pragma unroll
    for (int kc = 0; kc < 8; ++kc) {
      v16bf af = frag_from_lds_bf(xrow, kc * 32 + hi * 8);
#pragma unroll
      for (int i = 0; i < 4; ++i) {
        int nb = (w * 4 + i) * 16;
        acc[i] = wmma_bf(af, load_b_frag(W, nb + ln, kc * 32 + hi * 16), acc[i]);
      }
    }
  }

  uint4* dst = isA ? Afrag : Zfrag;
#pragma unroll
  for (int i = 0; i < 4; ++i) {
    int nt = w * 4 + i;
    uint4 o;
    o.x = pack_bf16(acc[i][0], acc[i][1]);
    o.y = pack_bf16(acc[i][2], acc[i][3]);
    o.z = pack_bf16(acc[i][4], acc[i][5]);
    o.w = pack_bf16(acc[i][6], acc[i][7]);
    dst[(((size_t)l0 * RT_COUNT + rt) * 16 + nt) * 32 + lane] = o;
  }
#if __has_builtin(__builtin_amdgcn_s_cluster_barrier)
  __builtin_amdgcn_s_cluster_barrier();   // NOP when not dispatched as a cluster
#endif
}

// ---------- kernel 2: 7-step recurrence, 16 chains per workgroup ----------
// s lives in registers (f32, one (m,n) owner per lane, matches WMMA C layout);
// bf16 shadow of s and h_c lives in LDS for zero-VALU A-fragment builds.
__global__ void __launch_bounds__(128) recurrence_kernel(
    const float* __restrict__ x, const float* __restrict__ end_inds,
    const float* __restrict__ bc, const u16* __restrict__ WB,
    const uint4* __restrict__ Afrag, const uint4* __restrict__ Zfrag,
    float* __restrict__ cum_ws, float* __restrict__ ssel) {
  __shared__ u16  s_bf[16 * STRB];
  __shared__ u16  hc_bf[16 * STRB];
  __shared__ float logits[16];
  __shared__ float cs_lds[16];
  __shared__ int   pos_lds[16];

  const int rt = blockIdx.x, rb = rt * 16, tid = threadIdx.x;
  const int w = tid >> 5, lane = tid & 31, ln = lane & 15, hi = lane >> 4;
  const u16* Wc1 = WB;                       // mat 0
  const u16* Wcs = WB + 2 * WMAT_ELEMS;      // mat 2

  // ---- init: s0 = x[:,0,:] into regs + bf16 LDS shadow ----
  float s_reg[4][8];
#pragma unroll
  for (int i = 0; i < 4; ++i) {
    int n = (w * 4 + i) * 16 + ln;
#pragma unroll
    for (int j = 0; j < 8; ++j) {
      int m = j + (hi ? 8 : 0);
      float v = x[(size_t)(rb + m) * LSEQ * FEAT + n];
      s_reg[i][j] = v;
      s_bf[m * STRB + n] = bf16_of(v);
    }
  }
  if (tid < 16) {
    int p = 1;
#pragma unroll
    for (int l = 1; l < LSEQ; ++l)
      if (end_inds[(size_t)(rb + tid) * LSEQ + l] > 0.5f) p = l;
    pos_lds[tid] = p;
  }
  float cum = 1.f, cumsel = 0.f;
  float bcv[4];
#pragma unroll
  for (int i = 0; i < 4; ++i) bcv[i] = bc[(w * 4 + i) * 16 + ln];

  v8f zero = {0.f, 0.f, 0.f, 0.f, 0.f, 0.f, 0.f, 0.f};

  for (int t = 0; t < LSEQ - 1; ++t) {
    if (tid < 16) logits[tid] = 0.f;
    __syncthreads();                         // s_bf + pos + logits-zero visible

    // ---- sW = s @ Wc1^T ----
    v16bf af[8];
#pragma unroll
    for (int kc = 0; kc < 8; ++kc) af[kc] = frag_from_lds_bf(&s_bf[ln * STRB], kc * 32 + hi * 8);
    v8f accc[4];
#pragma unroll
    for (int i = 0; i < 4; ++i) {
      v8f a = zero;
      int nb = (w * 4 + i) * 16;
#pragma unroll
      for (int kc = 0; kc < 8; ++kc)
        a = wmma_bf(af[kc], load_b_frag(Wc1, nb + ln, kc * 32 + hi * 16), a);
      accc[i] = a;
    }

    // ---- h_c (bf16 to LDS) and h_e (f32 regs) ----
    float he[4][8];
#pragma unroll
    for (int i = 0; i < 4; ++i) {
      int nt = w * 4 + i, n = nt * 16 + ln;
      size_t base = (((size_t)t * RT_COUNT + rt) * 16 + nt) * 32 + lane;
      uint4 zc = Zfrag[base];
      uint4 ze = Zfrag[base + (size_t)RT_COUNT * 16 * 32];   // l = t+1
      unsigned zcd[4] = {zc.x, zc.y, zc.z, zc.w};
      unsigned zed[4] = {ze.x, ze.y, ze.z, ze.w};
#pragma unroll
      for (int d = 0; d < 4; ++d) {
        int m0 = 2 * d + (hi ? 8 : 0), m1 = m0 + 1;
        float hc0 = accc[i][2 * d]     + bf_lo(zcd[d]) + bcv[i];
        float hc1 = accc[i][2 * d + 1] + bf_hi(zcd[d]) + bcv[i];
        hc_bf[m0 * STRB + n] = bf16_of(hc0);
        hc_bf[m1 * STRB + n] = bf16_of(hc1);
        he[i][2 * d]     = accc[i][2 * d]     + bf_lo(zed[d]) + bcv[i];
        he[i][2 * d + 1] = accc[i][2 * d + 1] + bf_hi(zed[d]) + bcv[i];
      }
    }
    __syncthreads();                         // hc_bf ready

    // ---- v = h_c @ Wcs^T ----
#pragma unroll
    for (int kc = 0; kc < 8; ++kc) af[kc] = frag_from_lds_bf(&hc_bf[ln * STRB], kc * 32 + hi * 8);
#pragma unroll
    for (int i = 0; i < 4; ++i) {
      v8f a = zero;
      int nb = (w * 4 + i) * 16;
#pragma unroll
      for (int kc = 0; kc < 8; ++kc)
        a = wmma_bf(af[kc], load_b_frag(Wcs, nb + ln, kc * 32 + hi * 16), a);
      accc[i] = a;                           // reuse as v
    }

    // ---- logits[m] = sum_n v*h_e  (LDS f32 atomics) ----
#pragma unroll
    for (int j = 0; j < 8; ++j) {
      float p = accc[0][j] * he[0][j] + accc[1][j] * he[1][j] +
                accc[2][j] * he[2][j] + accc[3][j] * he[3][j];
      atomicAdd(&logits[j + (hi ? 8 : 0)], p);
    }
    __syncthreads();                         // all atomics done

    if (tid < 16) {
      float csv = 1.f / (1.f + __expf(-logits[tid]));
      cs_lds[tid] = csv;
      cum *= csv;
      if (t == pos_lds[tid] - 1) cumsel = cum;
    }
    __syncthreads();                         // cs visible

    // ---- s = tanh(cs * A) + s ; refresh bf16 shadow; capture selected state ----
#pragma unroll
    for (int i = 0; i < 4; ++i) {
      int nt = w * 4 + i, n = nt * 16 + ln;
      uint4 a4 = Afrag[(((size_t)t * RT_COUNT + rt) * 16 + nt) * 32 + lane];
      unsigned ad[4] = {a4.x, a4.y, a4.z, a4.w};
#pragma unroll
      for (int d = 0; d < 4; ++d) {
#pragma unroll
        for (int h2 = 0; h2 < 2; ++h2) {
          int j = 2 * d + h2;
          int m = j + (hi ? 8 : 0);
          float av = h2 ? bf_hi(ad[d]) : bf_lo(ad[d]);
          float sn = tanhf(cs_lds[m] * av) + s_reg[i][j];
          s_reg[i][j] = sn;
          s_bf[m * STRB + n] = bf16_of(sn);
          if (t == pos_lds[m] - 1) ssel[(size_t)(rb + m) * FEAT + n] = sn;
        }
      }
      if (t < LSEQ - 2)
        __builtin_prefetch(&Afrag[(((size_t)(t + 1) * RT_COUNT + rt) * 16 + nt) * 32 + lane], 0, 0);
    }
    // next iteration's first barrier makes s_bf visible before reuse
  }
  if (tid < 16) cum_ws[rb + tid] = cumsel;
}

// ---------- kernel 3: softmax over 8 chains, mix, 512->2 output ----------
__global__ void __launch_bounds__(256) finalize_kernel(
    const float* __restrict__ x, const float* __restrict__ Wout,
    const float* __restrict__ bout, const float* __restrict__ cum_ws,
    const float* __restrict__ ssel, float* __restrict__ out) {
  __shared__ float wgt[8];
  __shared__ float r0[256], r1[256];
  const int n = blockIdx.x, tid = threadIdx.x;
  if (tid == 0) {
    float c[8], mx = -1e30f;
#pragma unroll
    for (int m = 0; m < 8; ++m) { c[m] = cum_ws[n * 8 + m]; mx = fmaxf(mx, c[m]); }
    float s = 0.f;
#pragma unroll
    for (int m = 0; m < 8; ++m) { c[m] = __expf(c[m] - mx); s += c[m]; }
#pragma unroll
    for (int m = 0; m < 8; ++m) wgt[m] = c[m] / s;
  }
  __syncthreads();
  const int d = tid;
  float u = 0.f;
#pragma unroll
  for (int m = 0; m < 8; ++m) u += wgt[m] * ssel[((size_t)n * 8 + m) * FEAT + d];
  float xv = x[(size_t)n * 8 * LSEQ * FEAT + d];       // x[8n, 0, d]
  r0[tid] = xv * Wout[d]            + u * Wout[FEAT + d];
  r1[tid] = xv * Wout[2 * FEAT + d] + u * Wout[3 * FEAT + d];
  __syncthreads();
  for (int sft = 128; sft > 0; sft >>= 1) {
    if (tid < sft) { r0[tid] += r0[tid + sft]; r1[tid] += r1[tid + sft]; }
    __syncthreads();
  }
  if (tid == 0) { out[2 * n] = r0[0] + bout[0]; out[2 * n + 1] = r1[0] + bout[1]; }
}

extern "C" void kernel_launch(void* const* d_in, const int* in_sizes, int n_in,
                              void* d_out, int out_size, void* d_ws, size_t ws_size,
                              hipStream_t stream) {
  const float* x    = (const float*)d_in[0];
  const float* ei   = (const float*)d_in[1];
  const float* Wc   = (const float*)d_in[3];
  const float* bc   = (const float*)d_in[4];
  const float* Wcs  = (const float*)d_in[5];
  const float* Want = (const float*)d_in[6];
  const float* Wout = (const float*)d_in[7];
  const float* bout = (const float*)d_in[8];
  float* out = (float*)d_out;

  unsigned char* ws = (unsigned char*)d_ws;
  u16* WB = (u16*)(ws + WB_OFF);
  uint4* Afrag  = (uint4*)(ws + A_OFF);
  uint4* Zfrag  = (uint4*)(ws + Z_OFF);
  float* cum_ws = (float*)(ws + CUM_OFF);
  float* ssel   = (float*)(ws + SSEL_OFF);

  prep_weights<<<(5 * WMAT_ELEMS + 255) / 256, 256, 0, stream>>>(Wc, Wcs, Want, WB);
  precompute_kernel<<<dim3(RT_COUNT, 15), 128, 0, stream>>>(x, WB, Afrag, Zfrag);
  recurrence_kernel<<<RT_COUNT, 128, 0, stream>>>(x, ei, bc, WB, Afrag, Zfrag, cum_ws, ssel);
  finalize_kernel<<<NGRP, 256, 0, stream>>>(x, Wout, bout, cum_ws, ssel, out);
}
